// Q_DenseUnit_44487271252340
// MI455X (gfx1250) — 8-bit matrix operand order repaired and checked by reading, compiled, not run on hardware
//
#include <hip/hip_runtime.h>
#include <cstdint>


#define BB   32
#define CIN  256
#define HH   56
#define WW   56
#define HWSZ 3136
#define NTOT 100352
#define BOT  128
#define GRW  32
#define COUT 288

typedef __attribute__((ext_vector_type(8))) int v8i;

static constexpr size_t OFF_SCAL = 0;
static constexpr size_t OFF_W1Q  = 64;
static constexpr size_t OFF_B1Q  = 1088;
static constexpr size_t OFF_SW1  = 2112;
static constexpr size_t OFF_BF   = 2624;
static constexpr size_t OFF_BFQ  = 3136;
static constexpr size_t OFF_SW2  = 3648;
static constexpr size_t OFF_QW1  = 4096;
static constexpr size_t OFF_QW2  = 36864;
static constexpr size_t OFF_YF   = 73728;
static constexpr size_t OFF_ZF   = OFF_YF + (size_t)BOT * NTOT * 4;


__device__ __forceinline__ int quant8(float x, float inv_s) {
  float q = rintf(x * inv_s);
  q = fminf(fmaxf(q, -128.f), 127.f);
  return (int)q;
}

__device__ __forceinline__ float block_max_256(float v, float* red) {
  int t = threadIdx.x;
  red[t] = v;
  __syncthreads();
  for (int s = 128; s > 0; s >>= 1) {
    if (t < s) red[t] = fmaxf(red[t], red[t + s]);
    __syncthreads();
  }
  return red[0];
}

__global__ __launch_bounds__(256) void k_prep(
    const float* __restrict__ g1, const float* __restrict__ be1,
    const float* __restrict__ mu1, const float* __restrict__ va1,
    const float* __restrict__ cw1,
    const float* __restrict__ g2, const float* __restrict__ be2,
    const float* __restrict__ mu2, const float* __restrict__ va2,
    const float* __restrict__ cw2, const float* __restrict__ s_in_p,
    float* scal, float* w1q, float* b1q, float* sw1, float* bf,
    int* qw1, float* sw2, int* qw2) {
  __shared__ float red[256];
  int t = threadIdx.x;
  if (t < 16) scal[t] = 0.f;

  float w1 = g1[t] * rsqrtf(va1[t] + 1e-5f);
  float b1 = be1[t] - mu1[t] * w1;
  float wmax = block_max_256(fabsf(w1), red);
  float s_bn = wmax / 127.f + 1e-8f;
  float qv = fminf(fmaxf(rintf(w1 / s_bn), -128.f), 127.f);
  w1q[t] = qv * s_bn;
  float sb = s_in_p[0] * s_bn;
  float qb = rintf(b1 / sb);
  qb = fminf(fmaxf(qb, -2147483649.f), 2147483648.f);
  b1q[t] = qb * sb;

  if (t < BOT) {
    float tt = g2[t] * rsqrtf(va2[t] + 1e-5f);
    const float* wrow = cw1 + (size_t)t * CIN;
    float mx = 0.f;
    for (int c = 0; c < CIN; ++c) mx = fmaxf(mx, fabsf(wrow[c] * tt));
    float sw = mx / 127.f + 1e-8f;
    sw1[t] = sw;
    float inv = 1.f / sw;
    for (int c = 0; c < CIN; c += 4) {
      int pk = 0;
      for (int j = 0; j < 4; ++j)
        pk |= (quant8(wrow[c + j] * tt, inv) & 255) << (8 * j);
      qw1[(t * CIN + c) >> 2] = pk;
    }
    bf[t] = be2[t] - mu2[t] * tt;
  }

  if (t < GRW) {
    const float* wrow = cw2 + (size_t)t * BOT * 9;
    float mx = 0.f;
    for (int i = 0; i < BOT * 9; ++i) mx = fmaxf(mx, fabsf(wrow[i]));
    float sw = mx / 127.f + 1e-8f;
    sw2[t] = sw;
    float inv = 1.f / sw;
    for (int tap = 0; tap < 9; ++tap)
      for (int c = 0; c < BOT; c += 4) {
        int pk = 0;
        for (int j = 0; j < 4; ++j)
          pk |= (quant8(wrow[(c + j) * 9 + tap], inv) & 255) << (8 * j);
        qw2[((t * 9 + tap) * BOT + c) >> 2] = pk;
      }
  }
}

__global__ __launch_bounds__(256) void k_act1_max(
    const float* __restrict__ batch, const float* __restrict__ w1q,
    const float* __restrict__ b1q, float* scal) {
  __shared__ float red[256];
  float m1 = 0.f, m2 = 0.f;
  const long long total = (long long)BB * CIN * HWSZ;
  for (long long i = (long long)blockIdx.x * 256 + threadIdx.x; i < total;
       i += (long long)gridDim.x * 256) {
    float v = batch[i];
    int c = (int)((i / HWSZ) % CIN);
    float x = fmaxf(v * w1q[c] + b1q[c], 0.f);
    m1 = fmaxf(m1, x);
    m2 = fmaxf(m2, fabsf(v));
  }
  float r1 = block_max_256(m1, red);
  __syncthreads();
  float r2 = block_max_256(m2, red);
  if (threadIdx.x == 0) {
    atomicMax((int*)&scal[0], __float_as_int(r1));
    atomicMax((int*)&scal[1], __float_as_int(r2));
  }
}

__global__ void k_fin1(float* scal, const float* bf, const float* sw1, float* bfq) {
  int t = threadIdx.x;
  float s_a1 = scal[0] / 127.f + 1e-8f;
  if (t == 0) scal[4] = s_a1;
  float s = s_a1 * sw1[t];
  float q = rintf(bf[t] / s);
  q = fminf(fmaxf(q, -2147483649.f), 2147483648.f);
  bfq[t] = q * s;
}

__global__ __launch_bounds__(256) void k_conv1(
    const float* __restrict__ batch, const float* __restrict__ scal,
    const float* __restrict__ w1q, const float* __restrict__ b1q,
    const int* __restrict__ qw1, const float* __restrict__ sw1,
    const float* __restrict__ bfq, float* __restrict__ yf, float* scal_out) {
  __shared__ __align__(16) int wl[BOT * CIN / 4];
  __shared__ __align__(16) int al[32 * CIN / 4];
  __shared__ float red[256];
  int t = threadIdx.x;

  for (int i = t; i < BOT * CIN / 4; i += 256) wl[i] = qw1[i];

  float s_a1 = scal[4];
  float inv_a1 = 1.f / s_a1;
  int n0 = blockIdx.x * 32;
  int b = n0 / HWSZ, sp0 = n0 % HWSZ;
  int col = t & 31, cg = t >> 5;
  const float* bp = batch + (size_t)b * CIN * HWSZ + sp0 + col;
  for (int c4 = 0; c4 < 8; ++c4) {
    int c = cg * 32 + c4 * 4;
    int pk = 0;
    for (int j = 0; j < 4; ++j) {
      float v = bp[(size_t)(c + j) * HWSZ];
      float x = fmaxf(v * w1q[c + j] + b1q[c + j], 0.f);
      pk |= (quant8(x, inv_a1) & 255) << (8 * j);
    }
    al[col * (CIN / 4) + (c >> 2)] = pk;
  }
  __syncthreads();

  int wv = t >> 5, l = t & 31;
  int m0 = wv * 16;
  int mrow = m0 + (l & 15);
  int hiA = (l < 16) ? 0 : 16;
  int hiB = (l < 16) ? 0 : 16;
  const char* wb = (const char*)wl;
  const char* ab = (const char*)al;
  v8i acc0 = {0, 0, 0, 0, 0, 0, 0, 0};
  v8i acc1 = {0, 0, 0, 0, 0, 0, 0, 0};

  for (int kk = 0; kk < CIN; kk += 64) {
    int baseA = mrow * CIN + kk + hiA;
    int2 a01 = *(const int2*)(wb + baseA);
    int2 a23 = *(const int2*)(wb + baseA + 8);
    int2 a45 = *(const int2*)(wb + baseA + 32);
    int2 a67 = *(const int2*)(wb + baseA + 40);
    v8i a;
    a[0] = a01.x; a[1] = a01.y; a[2] = a23.x; a[3] = a23.y;
    a[4] = a45.x; a[5] = a45.y; a[6] = a67.x; a[7] = a67.y;

    int baseB0 = (l & 15) * CIN + kk + hiB;
    int4 p0 = *(const int4*)(ab + baseB0);
    int4 p1 = *(const int4*)(ab + baseB0 + 32);
    v8i bb0;
    bb0[0] = p0.x; bb0[1] = p0.y; bb0[2] = p0.z; bb0[3] = p0.w;
    bb0[4] = p1.x; bb0[5] = p1.y; bb0[6] = p1.z; bb0[7] = p1.w;

    int baseB1 = (16 + (l & 15)) * CIN + kk + hiB;
    int4 p2 = *(const int4*)(ab + baseB1);
    int4 p3 = *(const int4*)(ab + baseB1 + 32);
    v8i bb1;
    bb1[0] = p2.x; bb1[1] = p2.y; bb1[2] = p2.z; bb1[3] = p2.w;
    bb1[4] = p3.x; bb1[5] = p3.y; bb1[6] = p3.z; bb1[7] = p3.w;

    acc0 = __builtin_amdgcn_wmma_i32_16x16x64_iu8(true, a, true, bb0, acc0, false, false);
    acc1 = __builtin_amdgcn_wmma_i32_16x16x64_iu8(true, a, true, bb1, acc1, false, false);
  }

  float tmax = 0.f;
  int nbase = n0 + (l & 15);
  for (int r = 0; r < 8; ++r) {
    int m = m0 + r + ((l < 16) ? 0 : 8);
    float sc = s_a1 * sw1[m];
    float bv = bfq[m];
    float y0 = fmaxf((float)acc0[r] * sc + bv, 0.f);
    float y1 = fmaxf((float)acc1[r] * sc + bv, 0.f);
    yf[(size_t)m * NTOT + nbase] = y0;
    yf[(size_t)m * NTOT + nbase + 16] = y1;
    tmax = fmaxf(tmax, fmaxf(y0, y1));
  }
  __syncthreads();
  float r0 = block_max_256(tmax, red);
  if (t == 0) atomicMax((int*)&scal_out[2], __float_as_int(r0));
}

__global__ void k_fin2(float* scal) {
  scal[5] = scal[2] / 127.f + 1e-8f;
}

__global__ __launch_bounds__(256) void k_conv2(
    const float* __restrict__ yf, const float* __restrict__ scal,
    const int* __restrict__ qw2, const float* __restrict__ sw2,
    float* __restrict__ zf, float* scal_out) {
  __shared__ __align__(16) int wl[GRW * 9 * BOT / 4];
  __shared__ __align__(16) int al[64 * BOT / 4];
  __shared__ float red[256];
  int t = threadIdx.x;

  for (int i = t; i < GRW * 9 * BOT / 4; i += 256) wl[i] = qw2[i];

  float s_a2 = scal[5];
  float inv_a2 = 1.f / s_a2;
  int n0 = blockIdx.x * 64;
  int b = n0 / HWSZ, sp0 = n0 % HWSZ;
  int col = t & 63, cq = t >> 6;
  int sp = sp0 + col;
  int h = sp / WW, w = sp % WW;

  int wv = t >> 5, l = t & 31;
  int m0 = (wv >> 2) * 16;
  int nb = (wv & 3) * 16;
  int mrow = m0 + (l & 15);
  int hiA = (l < 16) ? 0 : 16;
  int hiB = (l < 16) ? 0 : 16;
  const char* wb = (const char*)wl;
  const char* ab = (const char*)al;
  v8i acc = {0, 0, 0, 0, 0, 0, 0, 0};

  for (int tap = 0; tap < 9; ++tap) {
    int dh = tap / 3 - 1, dw = tap % 3 - 1;
    __syncthreads();
    int hh = h + dh, ww2 = w + dw;
    bool ok = (hh >= 0) & (hh < HH) & (ww2 >= 0) & (ww2 < WW);
    size_t srcsp = (size_t)b * HWSZ + hh * WW + ww2;
    for (int c4 = 0; c4 < 8; ++c4) {
      int c = cq * 32 + c4 * 4;
      int pk = 0;
      if (ok) {
        for (int j = 0; j < 4; ++j) {
          float v = yf[(size_t)(c + j) * NTOT + srcsp];
          pk |= (quant8(v, inv_a2) & 255) << (8 * j);
        }
      }
      al[col * (BOT / 4) + (c >> 2)] = pk;
    }
    __syncthreads();

    for (int kk = 0; kk < BOT; kk += 64) {
      int baseA = (mrow * 9 + tap) * BOT + kk + hiA;
      int2 a01 = *(const int2*)(wb + baseA);
      int2 a23 = *(const int2*)(wb + baseA + 8);
      int2 a45 = *(const int2*)(wb + baseA + 32);
      int2 a67 = *(const int2*)(wb + baseA + 40);
      v8i a;
      a[0] = a01.x; a[1] = a01.y; a[2] = a23.x; a[3] = a23.y;
      a[4] = a45.x; a[5] = a45.y; a[6] = a67.x; a[7] = a67.y;

      int baseB = (nb + (l & 15)) * BOT + kk + hiB;
      int4 p0 = *(const int4*)(ab + baseB);
      int4 p1 = *(const int4*)(ab + baseB + 32);
      v8i bb;
      bb[0] = p0.x; bb[1] = p0.y; bb[2] = p0.z; bb[3] = p0.w;
      bb[4] = p1.x; bb[5] = p1.y; bb[6] = p1.z; bb[7] = p1.w;

      acc = __builtin_amdgcn_wmma_i32_16x16x64_iu8(true, a, true, bb, acc, false, false);
    }
  }

  float tmax = 0.f;
  for (int r = 0; r < 8; ++r) {
    int m = m0 + r + ((l < 16) ? 0 : 8);
    float z = (float)acc[r] * (s_a2 * sw2[m]);
    zf[(size_t)m * NTOT + n0 + nb + (l & 15)] = z;
    tmax = fmaxf(tmax, fabsf(z));
  }
  __syncthreads();
  float r0 = block_max_256(tmax, red);
  if (t == 0) atomicMax((int*)&scal_out[3], __float_as_int(r0));
}

__global__ void k_fin3(float* scal) {
  scal[6] = fmaxf(scal[1], scal[3]) / 127.f + 1e-8f;
}

__global__ __launch_bounds__(256) void k_out(
    const float* __restrict__ batch, const float* __restrict__ zf,
    const float* __restrict__ scal, float* __restrict__ out) {
  float s = scal[6];
  float inv = 1.f / s;
  const long long total = (long long)BB * COUT * HWSZ;
  for (long long i = (long long)blockIdx.x * 256 + threadIdx.x; i < total;
       i += (long long)gridDim.x * 256) {
    long long sp = i % HWSZ;
    long long c = (i / HWSZ) % COUT;
    long long b = i / ((long long)COUT * HWSZ);
    float v;
    if (c < CIN) v = batch[(b * CIN + c) * HWSZ + sp];
    else         v = zf[(c - CIN) * (long long)NTOT + b * HWSZ + sp];
    float q = fminf(fmaxf(rintf(v * inv), -128.f), 127.f);
    out[i] = q * s;
  }
  if (blockIdx.x == 0 && threadIdx.x == 0) out[total] = s;
}

extern "C" void kernel_launch(void* const* d_in, const int* in_sizes, int n_in,
                              void* d_out, int out_size, void* d_ws, size_t ws_size,
                              hipStream_t stream) {
  const float* batch = (const float*)d_in[0];
  const float* s_in  = (const float*)d_in[1];
  const float* g1    = (const float*)d_in[2];
  const float* be1   = (const float*)d_in[3];
  const float* mu1   = (const float*)d_in[4];
  const float* va1   = (const float*)d_in[5];
  const float* cw1   = (const float*)d_in[6];
  const float* g2    = (const float*)d_in[7];
  const float* be2   = (const float*)d_in[8];
  const float* mu2   = (const float*)d_in[9];
  const float* va2   = (const float*)d_in[10];
  const float* cw2   = (const float*)d_in[11];
  float* out = (float*)d_out;

  char* ws = (char*)d_ws;
  float* scal = (float*)(ws + OFF_SCAL);
  float* w1q  = (float*)(ws + OFF_W1Q);
  float* b1q  = (float*)(ws + OFF_B1Q);
  float* sw1  = (float*)(ws + OFF_SW1);
  float* bf   = (float*)(ws + OFF_BF);
  float* bfq  = (float*)(ws + OFF_BFQ);
  float* sw2  = (float*)(ws + OFF_SW2);
  int*   qw1  = (int*)(ws + OFF_QW1);
  int*   qw2  = (int*)(ws + OFF_QW2);
  float* yf   = (float*)(ws + OFF_YF);
  float* zf   = (float*)(ws + OFF_ZF);

  k_prep<<<1, 256, 0, stream>>>(g1, be1, mu1, va1, cw1, g2, be2, mu2, va2,
                                cw2, s_in, scal, w1q, b1q, sw1, bf, qw1, sw2, qw2);
  k_act1_max<<<2048, 256, 0, stream>>>(batch, w1q, b1q, scal);
  k_fin1<<<1, 128, 0, stream>>>(scal, bf, sw1, bfq);
  k_conv1<<<NTOT / 32, 256, 0, stream>>>(batch, scal, w1q, b1q, qw1, sw1, bfq, yf, scal);
  k_fin2<<<1, 1, 0, stream>>>(scal);
  k_conv2<<<NTOT / 64, 256, 0, stream>>>(yf, scal, qw2, sw2, zf, scal);
  k_fin3<<<1, 1, 0, stream>>>(scal);
  k_out<<<4096, 256, 0, stream>>>(batch, zf, scal, out);
}
